// GNN_27530740367556
// MI455X (gfx1250) — compile-verified, matrix operand order checked by reading, not run on hardware
//
#include <hip/hip_runtime.h>
#include <hip/hip_bf16.h>
#include <math.h>

// ---------------- problem constants (match reference) ----------------
#define N_NODES  100000
#define N_EDGES  1600000
#define HIDDEN   128
#define N_GRAPHS 256
#define L1DIM    64

// ---------------- CDNA5 WMMA vector types ----------------
typedef __bf16 bf16_t;
typedef __attribute__((ext_vector_type(16))) __bf16 v16bf;
typedef __attribute__((ext_vector_type(8)))  __bf16 v8bf;
typedef __attribute__((ext_vector_type(8)))  float  v8f;

// hardware RNE conversion: lowers to v_cvt_pk_bf16_f32 when pair-vectorized
__device__ __forceinline__ bf16_t to_bf16(float f) { return (bf16_t)f; }

// A-fragment (16x32 bf16, ISA 7.12.2 layout): lane holds row M=lane&15,
// kbase = lane<16 ? 0 : 8; elem j<8 -> K=kbase+j, elem j>=8 -> K=kbase+16+(j-8).
__device__ __forceinline__ v16bf load_a_frag_f32(const float* __restrict__ row, int k0) {
    const float4 r0 = *(const float4*)(row + k0);
    const float4 r1 = *(const float4*)(row + k0 + 4);
    const float4 r2 = *(const float4*)(row + k0 + 16);
    const float4 r3 = *(const float4*)(row + k0 + 20);
    float t[16] = { r0.x, r0.y, r0.z, r0.w, r1.x, r1.y, r1.z, r1.w,
                    r2.x, r2.y, r2.z, r2.w, r3.x, r3.y, r3.z, r3.w };
    v16bf a;
#pragma unroll
    for (int j = 0; j < 16; ++j) a[j] = to_bf16(t[j]);
    return a;
}

// B-fragment (32x16 bf16): lane holds column N=lane&15 = weight row; same K map.
__device__ __forceinline__ v16bf load_b_frag(const bf16_t* __restrict__ wrow, int k0) {
    const v8bf b0 = *(const v8bf*)(wrow + k0);
    const v8bf b1 = *(const v8bf*)(wrow + k0 + 16);
    v16bf b;
#pragma unroll
    for (int j = 0; j < 8; ++j) { b[j] = b0[j]; b[8 + j] = b1[j]; }
    return b;
}

// ---------------- kernels ----------------

__global__ void zero_f32_kernel(float* __restrict__ p, int n) {
    int i = blockIdx.x * blockDim.x + threadIdx.x;
    if (i < n) p[i] = 0.0f;
}

// convert 4 [128x128] f32 weight matrices to bf16 (packed consecutively)
__global__ void convert_w_kernel(const float* __restrict__ w0, const float* __restrict__ w1,
                                 const float* __restrict__ w2, const float* __restrict__ w3,
                                 bf16_t* __restrict__ dst) {
    int i = blockIdx.x * blockDim.x + threadIdx.x;
    if (i >= 4 * HIDDEN * HIDDEN) return;
    const float* srcs[4] = { w0, w1, w2, w3 };
    dst[i] = to_bf16(srcs[i >> 14][i & 16383]);
}

// embedding gather + max_norm=1 renorm; one wave32 per node, float4 per lane
__global__ void embed_kernel(const int* __restrict__ x, const float* __restrict__ emb,
                             float* __restrict__ h) {
    int n = (blockIdx.x * blockDim.x + threadIdx.x) >> 5;
    if (n >= N_NODES) return;
    int lane = threadIdx.x & 31;
    int row = x[n];
    float4 v = ((const float4*)(emb + (size_t)row * HIDDEN))[lane];
    float ss = v.x * v.x + v.y * v.y + v.z * v.z + v.w * v.w;
#pragma unroll
    for (int off = 16; off > 0; off >>= 1) ss += __shfl_xor(ss, off, 32);
    float norm  = sqrtf(ss);
    float scale = fminf(1.0f, 1.0f / (norm + 1e-7f));
    float4 o = { v.x * scale, v.y * scale, v.z * scale, v.w * scale };
    ((float4*)(h + (size_t)n * HIDDEN))[lane] = o;
}

// per-edge message + scatter-add: one wave32 per edge, hardware f32 atomics
__global__ void scatter_kernel(const int* __restrict__ src, const int* __restrict__ dst,
                               const float* __restrict__ ew,
                               const float* __restrict__ h, float* __restrict__ agg) {
    int e = (blockIdx.x * blockDim.x + threadIdx.x) >> 5;
    if (e >= N_EDGES) return;
    int lane = threadIdx.x & 31;
    int s = src[e], d = dst[e];
    float w = ew[e];
    float4 v = ((const float4*)(h + (size_t)s * HIDDEN))[lane];
    float* ap = agg + (size_t)d * HIDDEN + lane * 4;
    __hip_atomic_fetch_add(ap + 0, v.x * w, __ATOMIC_RELAXED, __HIP_MEMORY_SCOPE_AGENT);
    __hip_atomic_fetch_add(ap + 1, v.y * w, __ATOMIC_RELAXED, __HIP_MEMORY_SCOPE_AGENT);
    __hip_atomic_fetch_add(ap + 2, v.z * w, __ATOMIC_RELAXED, __HIP_MEMORY_SCOPE_AGENT);
    __hip_atomic_fetch_add(ap + 3, v.w * w, __ATOMIC_RELAXED, __HIP_MEMORY_SCOPE_AGENT);
}

// out = relu(agg @ Wrel^T + bias + h @ Wroot^T); one wave per 16-node tile.
// bf16 WMMA with f32 accumulation. Writes in-place over agg-buffer rows safely
// (each wave reads its own tile fully before storing).
__global__ __launch_bounds__(256) void linear_wmma_kernel(
    const float* __restrict__ agg, const float* __restrict__ h,
    const bf16_t* __restrict__ Wrel, const bf16_t* __restrict__ Wroot,
    const float* __restrict__ bias, float* __restrict__ out, int nTiles) {
    int wave = (blockIdx.x * blockDim.x + threadIdx.x) >> 5;
    if (wave >= nTiles) return;
    int lane  = threadIdx.x & 31;
    int m     = lane & 15;                 // A row / B column / D column
    int kbase = (lane < 16) ? 0 : 8;
    int node  = wave * 16 + m;
    const float* aggRow = agg + (size_t)node * HIDDEN;
    const float* hRow   = h   + (size_t)node * HIDDEN;

    v16bf aA[4], aH[4];
#pragma unroll
    for (int kc = 0; kc < 4; ++kc) {
        aA[kc] = load_a_frag_f32(aggRow, kc * 32 + kbase);
        aH[kc] = load_a_frag_f32(hRow,   kc * 32 + kbase);
    }

    int mhi = (lane < 16) ? 0 : 8;
#pragma unroll
    for (int ot = 0; ot < 8; ++ot) {
        float bv = bias[ot * 16 + m];
        v8f acc = { bv, bv, bv, bv, bv, bv, bv, bv };
        const bf16_t* wrel  = Wrel  + (size_t)(ot * 16 + m) * HIDDEN;
        const bf16_t* wroot = Wroot + (size_t)(ot * 16 + m) * HIDDEN;
#pragma unroll
        for (int kc = 0; kc < 4; ++kc) {
            v16bf bR = load_b_frag(wrel,  kc * 32 + kbase);
            acc = __builtin_amdgcn_wmma_f32_16x16x32_bf16(
                false, aA[kc], false, bR, (short)0, acc, false, false);
            v16bf bT = load_b_frag(wroot, kc * 32 + kbase);
            acc = __builtin_amdgcn_wmma_f32_16x16x32_bf16(
                false, aH[kc], false, bT, (short)0, acc, false, false);
        }
#pragma unroll
        for (int v = 0; v < 8; ++v) {
            float xv = acc[v];
            xv = xv > 0.0f ? xv : 0.0f;    // relu
            int nodeD = wave * 16 + v + mhi;
            out[(size_t)nodeD * HIDDEN + ot * 16 + m] = xv;
        }
    }
}

// mean-pool accumulation: one wave per node
__global__ void pool_kernel(const float* __restrict__ h, const int* __restrict__ batch,
                            float* __restrict__ sums, float* __restrict__ counts) {
    int n = (blockIdx.x * blockDim.x + threadIdx.x) >> 5;
    if (n >= N_NODES) return;
    int lane = threadIdx.x & 31;
    int g = batch[n];
    float4 v = ((const float4*)(h + (size_t)n * HIDDEN))[lane];
    float* sp = sums + (size_t)g * HIDDEN + lane * 4;
    __hip_atomic_fetch_add(sp + 0, v.x, __ATOMIC_RELAXED, __HIP_MEMORY_SCOPE_AGENT);
    __hip_atomic_fetch_add(sp + 1, v.y, __ATOMIC_RELAXED, __HIP_MEMORY_SCOPE_AGENT);
    __hip_atomic_fetch_add(sp + 2, v.z, __ATOMIC_RELAXED, __HIP_MEMORY_SCOPE_AGENT);
    __hip_atomic_fetch_add(sp + 3, v.w, __ATOMIC_RELAXED, __HIP_MEMORY_SCOPE_AGENT);
    if (lane == 0)
        __hip_atomic_fetch_add(counts + g, 1.0f, __ATOMIC_RELAXED, __HIP_MEMORY_SCOPE_AGENT);
}

// tiny MLP head + softmax: one thread per graph
__global__ void classifier_kernel(const float* __restrict__ sums, const float* __restrict__ counts,
                                  const float* __restrict__ w1, const float* __restrict__ b1,
                                  const float* __restrict__ w2, const float* __restrict__ b2,
                                  float* __restrict__ out) {
    int g = threadIdx.x;
    if (g >= N_GRAPHS) return;
    float inv = 1.0f / fmaxf(counts[g], 1.0f);
    const float* p = sums + (size_t)g * HIDDEN;
    float z1[L1DIM];
    for (int o = 0; o < L1DIM; ++o) {
        float s = b1[o];
        const float* wr = w1 + (size_t)o * HIDDEN;
        for (int k = 0; k < HIDDEN; ++k) s += p[k] * inv * wr[k];
        z1[o] = s > 0.0f ? s : 0.0f;
    }
    float z2[2];
    for (int c = 0; c < 2; ++c) {
        float s = b2[c];
        for (int o = 0; o < L1DIM; ++o) s += z1[o] * w2[c * L1DIM + o];
        z2[c] = s;
    }
    float mx = fmaxf(z2[0], z2[1]);
    float e0 = __expf(z2[0] - mx), e1 = __expf(z2[1] - mx);
    float inv2 = 1.0f / (e0 + e1);
    out[g * 2 + 0] = e0 * inv2;
    out[g * 2 + 1] = e1 * inv2;
}

// ---------------- launch ----------------
extern "C" void kernel_launch(void* const* d_in, const int* in_sizes, int n_in,
                              void* d_out, int out_size, void* d_ws, size_t ws_size,
                              hipStream_t stream) {
    (void)in_sizes; (void)n_in; (void)out_size; (void)ws_size;

    const int*   x          = (const int*)d_in[0];
    const int*   edge_index = (const int*)d_in[1];
    const float* edge_w     = (const float*)d_in[2];
    const int*   batch      = (const int*)d_in[3];
    const float* emb_w      = (const float*)d_in[4];
    const float* w1_rel     = (const float*)d_in[5];
    const float* b1_rel     = (const float*)d_in[6];
    const float* w1_root    = (const float*)d_in[7];
    const float* w2_rel     = (const float*)d_in[8];
    const float* b2_rel     = (const float*)d_in[9];
    const float* w2_root    = (const float*)d_in[10];
    const float* cls1_w     = (const float*)d_in[11];
    const float* cls1_b     = (const float*)d_in[12];
    const float* cls2_w     = (const float*)d_in[13];
    const float* cls2_b     = (const float*)d_in[14];

    const int* e_src = edge_index;
    const int* e_dst = edge_index + N_EDGES;

    // workspace layout (all offsets are multiples of 256 B down to the bf16 block)
    float*  hA     = (float*)d_ws;                          // [N, 128]
    float*  hB     = hA + (size_t)N_NODES * HIDDEN;         // [N, 128]
    float*  pooled = hB + (size_t)N_NODES * HIDDEN;         // [256, 128]
    float*  counts = pooled + (size_t)N_GRAPHS * HIDDEN;    // [256]
    bf16_t* wbf    = (bf16_t*)(counts + N_GRAPHS);          // 4 x [128*128] bf16
    bf16_t* w1rel_bf  = wbf + 0 * HIDDEN * HIDDEN;
    bf16_t* w1root_bf = wbf + 1 * HIDDEN * HIDDEN;
    bf16_t* w2rel_bf  = wbf + 2 * HIDDEN * HIDDEN;
    bf16_t* w2root_bf = wbf + 3 * HIDDEN * HIDDEN;

    const int nFeat      = N_NODES * HIDDEN;                // 12.8M
    const int nTiles     = N_NODES / 16;                    // 6250
    const int nodeBlocks = (N_NODES * 32 + 255) / 256;      // wave per node
    const int edgeBlocks = (N_EDGES * 32 + 255) / 256;      // wave per edge
    const int gemmBlocks = (nTiles + 7) / 8;                // 8 waves/block

    // weights -> bf16
    convert_w_kernel<<<(4 * HIDDEN * HIDDEN + 255) / 256, 256, 0, stream>>>(
        w1_rel, w1_root, w2_rel, w2_root, wbf);

    // h0 = renormed embedding
    embed_kernel<<<nodeBlocks, 256, 0, stream>>>(x, emb_w, hA);

    // layer 1: hB = relu(scatter(hA) @ W1rel^T + b1 + hA @ W1root^T)
    zero_f32_kernel<<<(nFeat + 255) / 256, 256, 0, stream>>>(hB, nFeat);
    scatter_kernel<<<edgeBlocks, 256, 0, stream>>>(e_src, e_dst, edge_w, hA, hB);
    linear_wmma_kernel<<<gemmBlocks, 256, 0, stream>>>(
        hB, hA, w1rel_bf, w1root_bf, b1_rel, hB, nTiles);

    // layer 2: hA = relu(scatter(hB) @ W2rel^T + b2 + hB @ W2root^T)
    zero_f32_kernel<<<(nFeat + 255) / 256, 256, 0, stream>>>(hA, nFeat);
    scatter_kernel<<<edgeBlocks, 256, 0, stream>>>(e_src, e_dst, edge_w, hB, hA);
    linear_wmma_kernel<<<gemmBlocks, 256, 0, stream>>>(
        hA, hB, w2rel_bf, w2root_bf, b2_rel, hA, nTiles);

    // global mean pool
    zero_f32_kernel<<<(N_GRAPHS * HIDDEN + N_GRAPHS + 255) / 256, 256, 0, stream>>>(
        pooled, N_GRAPHS * HIDDEN + N_GRAPHS);
    pool_kernel<<<nodeBlocks, 256, 0, stream>>>(hA, batch, pooled, counts);

    // classifier + softmax
    classifier_kernel<<<1, 256, 0, stream>>>(
        pooled, counts, cls1_w, cls1_b, cls2_w, cls2_b, (float*)d_out);
}
